// KoLeoLossViewExpanded_14654428414600
// MI455X (gfx1250) — matrix operand order repaired and checked by reading, not compiled, not run on hardware
//
#include <hip/hip_runtime.h>
#include <hip/hip_bf16.h>

// ---------------------------------------------------------------------------
// KoLeo loss, fused for CDNA5 (gfx1250, wave32, WMMA).
//   x: [8192, 4, 1024] fp32.  Per view t: normalize rows, NN via argmax of
//   Xn@Xn^T (diag masked), loss = -mean(log(||xn - xn[I] + eps|| + eps)).
// Gram/argmax phase: v_wmma_f32_16x16x32_bf16, A panel staged to LDS with
// async global->LDS copies, B double-buffered in registers. Distances are
// recomputed in fp32, final reduction is a deterministic tree.
// ---------------------------------------------------------------------------

typedef __attribute__((ext_vector_type(16))) __bf16 v16bf;
typedef __attribute__((ext_vector_type(8)))  float  v8f;

#define B_ROWS      8192
#define T_VIEWS     4
#define D_DIM       1024
#define NROWS       (B_ROWS * T_VIEWS)   // 32768
#define ROWS_PER_WG 32                   // A panel rows per workgroup

static __device__ __forceinline__ unsigned short f32_to_bf16_rne(float f) {
    unsigned int u = __float_as_uint(f);
    if ((u & 0x7fffffffu) > 0x7f800000u)       // NaN: quiet it
        return (unsigned short)((u >> 16) | 0x0040u);
    u += 0x7fffu + ((u >> 16) & 1u);           // round-to-nearest-even
    return (unsigned short)(u >> 16);
}

// ---------------------------------------------------------------------------
// Kernel 1: per-row L2 norm; store bf16 normalized rows (view-major) + 1/norm.
// ---------------------------------------------------------------------------
__global__ void __launch_bounds__(256)
koleo_normalize(const float* __restrict__ x,
                unsigned short* __restrict__ xnbf,
                float* __restrict__ invn) {
    const int rowlin = blockIdx.x;             // view-major: t*8192 + b
    const int t = rowlin >> 13;
    const int b = rowlin & (B_ROWS - 1);
    const float* row = x + ((size_t)b * T_VIEWS + t) * D_DIM;

    const int tid  = threadIdx.x;
    const int lane = tid & 31;
    const int wid  = tid >> 5;

    float v[4];
    float ss = 0.f;
#pragma unroll
    for (int i = 0; i < 4; ++i) {
        v[i] = row[tid + i * 256];
        ss += v[i] * v[i];
    }
#pragma unroll
    for (int off = 16; off > 0; off >>= 1) ss += __shfl_xor(ss, off, 32);

    __shared__ float red[8];
    if (lane == 0) red[wid] = ss;
    __syncthreads();
    if (tid == 0) {
        float tot = 0.f;
        for (int i = 0; i < 8; ++i) tot += red[i];
        red[0] = tot;
    }
    __syncthreads();
    const float inv = 1.0f / fmaxf(sqrtf(red[0]), 1e-12f);

    unsigned short* orow = xnbf + (size_t)rowlin * D_DIM;
#pragma unroll
    for (int i = 0; i < 4; ++i)
        orow[tid + i * 256] = f32_to_bf16_rne(v[i] * inv);
    if (tid == 0) invn[rowlin] = inv;
}

// ---------------------------------------------------------------------------
// Kernel 2: fused Gram-matrix + row argmax via v_wmma_f32_16x16x32_bf16.
// Block = 8 waves (256 thr), owns 32 rows of one view. 64 column iterations
// (8 waves x 16 cols each). Per k-step each wave does 2 WMMAs (two 16-row
// tiles) sharing one register-double-buffered B fragment; A fragments come
// from a 64KB LDS panel staged once with async global->LDS b128 copies.
// ---------------------------------------------------------------------------
__global__ void __launch_bounds__(256)
koleo_argmax_gemm(const unsigned short* __restrict__ xnbf,
                  int* __restrict__ nn_idx) {
    __shared__ unsigned short shA[ROWS_PER_WG * D_DIM];   // 64 KB A panel
    __shared__ float smax[8][ROWS_PER_WG];
    __shared__ int   sidx[8][ROWS_PER_WG];

    const int view    = blockIdx.x >> 8;                  // 256 row-blocks/view
    const int rowbase = (blockIdx.x & 255) * ROWS_PER_WG;
    const int tid  = threadIdx.x;
    const int lane = tid & 31;
    const int wid  = tid >> 5;

    const unsigned short* xview = xnbf + (size_t)view * B_ROWS * D_DIM;

    // ---- Stage A panel (contiguous 64KB) via async global->LDS copies ----
    {
        const char* src = (const char*)(xview + (size_t)rowbase * D_DIM);
        const unsigned ldsbase = (unsigned)(size_t)(&shA[0]);
#pragma unroll
        for (int i = 0; i < 16; ++i) {
            const unsigned boff     = (unsigned)(tid + i * 256) * 16u;
            const unsigned lds_addr = ldsbase + boff;
            const unsigned long long gaddr = (unsigned long long)(size_t)(src + boff);
            asm volatile("global_load_async_to_lds_b128 %0, %1, off"
                         :: "v"(lds_addr), "v"(gaddr) : "memory");
        }
        asm volatile("s_wait_asynccnt 0x0" ::: "memory");
    }
    __syncthreads();

    const int nloc  = lane & 15;   // C: column within tile; A: row M
    const int mhalf = lane >> 4;   // A/B: K-group select; C: row-half select

    float best0[8], best1[8];
    int   bidx0[8], bidx1[8];
#pragma unroll
    for (int r = 0; r < 8; ++r) {
        best0[r] = -3.0f; bidx0[r] = 0;
        best1[r] = -3.0f; bidx1[r] = 0;
    }

    union Frag { v16bf v; uint4 q[2]; };

    // A fragment sources: row = lane&15 (+16 for tile 1),
// K elems {k0+mhalf*16..+15}, contiguous 32B.
    const unsigned short* arow0 = shA + (size_t)(lane & 15) * D_DIM + mhalf * 16;
    const unsigned short* arow1 = arow0 + 16 * D_DIM;

    for (int it = 0; it < 64; ++it) {
        const int ncol = it * 128 + wid * 16 + nloc;   // global column index
        // B fragment: lane n holds K = mhalf*16 + {0..15}, contiguous 32B.
        const uint4* brow = reinterpret_cast<const uint4*>(
            xview + (size_t)ncol * D_DIM + mhalf * 16);

        v8f c0 = {}, c1 = {};
        Frag b0, b1;
        b0.q[0] = brow[0]; b0.q[1] = brow[1];          // k=0 fragment

#pragma unroll 2
        for (int k0 = 0; k0 < D_DIM; k0 += 64) {
            Frag a0, a1;
            // ---- k0: consume b0, prefetch b1 (k0+32) ----
            b1.q[0] = brow[4]; b1.q[1] = brow[5];
            {
                const uint4* ap0 = reinterpret_cast<const uint4*>(arow0 + k0);
                const uint4* ap1 = reinterpret_cast<const uint4*>(arow1 + k0);
                a0.q[0] = ap0[0]; a0.q[1] = ap0[1];
                a1.q[0] = ap1[0]; a1.q[1] = ap1[1];
            }
            c0 = __builtin_amdgcn_wmma_f32_16x16x32_bf16(
                     false, a0.v, false, b0.v, (short)0, c0, false, false);
            c1 = __builtin_amdgcn_wmma_f32_16x16x32_bf16(
                     false, a1.v, false, b0.v, (short)0, c1, false, false);
            // ---- k0+32: consume b1, prefetch b0 (k0+64; last one overreads
            //      32B into the following ws arrays, never consumed) ----
            b0.q[0] = brow[8]; b0.q[1] = brow[9];
            {
                const uint4* ap0 = reinterpret_cast<const uint4*>(arow0 + k0 + 32);
                const uint4* ap1 = reinterpret_cast<const uint4*>(arow1 + k0 + 32);
                a0.q[0] = ap0[0]; a0.q[1] = ap0[1];
                a1.q[0] = ap1[0]; a1.q[1] = ap1[1];
            }
            c0 = __builtin_amdgcn_wmma_f32_16x16x32_bf16(
                     false, a0.v, false, b1.v, (short)0, c0, false, false);
            c1 = __builtin_amdgcn_wmma_f32_16x16x32_bf16(
                     false, a1.v, false, b1.v, (short)0, c1, false, false);
            brow += 8;
        }

        // C layout: lane n = nloc, VGPR r holds M = mhalf*8 + r (tile0),
        // M = 16 + mhalf*8 + r (tile1).
#pragma unroll
        for (int r = 0; r < 8; ++r) {
            const int mg0 = rowbase + mhalf * 8 + r;
            float v0 = c0[r];
            if (ncol == mg0)      v0 = -1.0f;                  // mask diagonal
            if (v0 > best0[r]) { best0[r] = v0; bidx0[r] = ncol; }
            float v1 = c1[r];
            if (ncol == mg0 + 16) v1 = -1.0f;
            if (v1 > best1[r]) { best1[r] = v1; bidx1[r] = ncol; }
        }
    }

    // Reduce across the 16 lanes of each half (xor masks < 16 stay in-half).
#pragma unroll
    for (int r = 0; r < 8; ++r) {
        float bv0 = best0[r]; int bi0 = bidx0[r];
        float bv1 = best1[r]; int bi1 = bidx1[r];
#pragma unroll
        for (int off = 8; off > 0; off >>= 1) {
            const float ov0 = __shfl_xor(bv0, off, 32);
            const int   oi0 = __shfl_xor(bi0, off, 32);
            if (ov0 > bv0 || (ov0 == bv0 && oi0 < bi0)) { bv0 = ov0; bi0 = oi0; }
            const float ov1 = __shfl_xor(bv1, off, 32);
            const int   oi1 = __shfl_xor(bi1, off, 32);
            if (ov1 > bv1 || (ov1 == bv1 && oi1 < bi1)) { bv1 = ov1; bi1 = oi1; }
        }
        best0[r] = bv0; bidx0[r] = bi0;
        best1[r] = bv1; bidx1[r] = bi1;
    }
    if (nloc == 0) {               // lanes 0 (rows 0..7) and 16 (rows 8..15)
#pragma unroll
        for (int r = 0; r < 8; ++r) {
            smax[wid][mhalf * 8 + r]      = best0[r];
            sidx[wid][mhalf * 8 + r]      = bidx0[r];
            smax[wid][16 + mhalf * 8 + r] = best1[r];
            sidx[wid][16 + mhalf * 8 + r] = bidx1[r];
        }
    }
    __syncthreads();

    // Combine the 8 waves' column ranges; smallest index wins ties.
    if (tid < ROWS_PER_WG) {
        float bv = smax[0][tid]; int bi = sidx[0][tid];
        for (int w = 1; w < 8; ++w) {
            const float ov = smax[w][tid]; const int oi = sidx[w][tid];
            if (ov > bv || (ov == bv && oi < bi)) { bv = ov; bi = oi; }
        }
        nn_idx[view * B_ROWS + rowbase + tid] = bi;
    }
}

// ---------------------------------------------------------------------------
// Kernel 3: per-row loss in fp32 (renormalize from raw input + stored 1/norm).
// ---------------------------------------------------------------------------
__global__ void __launch_bounds__(256)
koleo_rowloss(const float* __restrict__ x,
              const float* __restrict__ invn,
              const int*   __restrict__ nn_idx,
              float*       __restrict__ rowloss) {
    const int rowlin = blockIdx.x;
    const int t  = rowlin >> 13;
    const int b  = rowlin & (B_ROWS - 1);
    const int nb = nn_idx[rowlin];

    const float* xm = x + ((size_t)b  * T_VIEWS + t) * D_DIM;
    const float* xi = x + ((size_t)nb * T_VIEWS + t) * D_DIM;
    const float sm = invn[rowlin];
    const float si = invn[t * B_ROWS + nb];

    const int tid  = threadIdx.x;
    const int lane = tid & 31;
    const int wid  = tid >> 5;

    float ss = 0.f;
#pragma unroll
    for (int i = 0; i < 4; ++i) {
        const int k = tid + i * 256;
        const float d = xm[k] * sm - xi[k] * si + 1e-12f;
        ss += d * d;
    }
#pragma unroll
    for (int off = 16; off > 0; off >>= 1) ss += __shfl_xor(ss, off, 32);

    __shared__ float red[8];
    if (lane == 0) red[wid] = ss;
    __syncthreads();
    if (tid == 0) {
        float tot = 0.f;
        for (int i = 0; i < 8; ++i) tot += red[i];
        rowloss[rowlin] = -logf(sqrtf(tot) + 1e-12f);
    }
}

// ---------------------------------------------------------------------------
// Kernel 4: deterministic tree reduction of 32768 row losses -> scalar mean.
// ---------------------------------------------------------------------------
__global__ void __launch_bounds__(256)
koleo_finalize(const float* __restrict__ rowloss, float* __restrict__ out) {
    const int tid  = threadIdx.x;
    const int lane = tid & 31;
    const int wid  = tid >> 5;

    float s = 0.f;
    for (int i = tid; i < NROWS; i += 256) s += rowloss[i];
#pragma unroll
    for (int off = 16; off > 0; off >>= 1) s += __shfl_xor(s, off, 32);

    __shared__ float red[8];
    if (lane == 0) red[wid] = s;
    __syncthreads();
    if (tid == 0) {
        float tot = 0.f;
        for (int i = 0; i < 8; ++i) tot += red[i];
        out[0] = tot / (float)NROWS;   // mean over views == mean over all rows
    }
}

// ---------------------------------------------------------------------------
extern "C" void kernel_launch(void* const* d_in, const int* in_sizes, int n_in,
                              void* d_out, int out_size, void* d_ws, size_t ws_size,
                              hipStream_t stream) {
    (void)in_sizes; (void)n_in; (void)out_size; (void)ws_size;

    const float* x   = (const float*)d_in[0];
    float*       out = (float*)d_out;

    char* ws = (char*)d_ws;
    unsigned short* xnbf = (unsigned short*)ws;                       // 64 MB bf16 Xn (view-major)
    size_t off = (size_t)NROWS * D_DIM * sizeof(unsigned short);
    float* invn    = (float*)(ws + off); off += (size_t)NROWS * sizeof(float);
    int*   nn      = (int*)  (ws + off); off += (size_t)NROWS * sizeof(int);
    float* rowloss = (float*)(ws + off);

    koleo_normalize  <<<dim3(NROWS), dim3(256), 0, stream>>>(x, xnbf, invn);
    koleo_argmax_gemm<<<dim3(T_VIEWS * (B_ROWS / ROWS_PER_WG)), dim3(256), 0, stream>>>(xnbf, nn);
    koleo_rowloss    <<<dim3(NROWS), dim3(256), 0, stream>>>(x, invn, nn, rowloss);
    koleo_finalize   <<<dim3(1),     dim3(256), 0, stream>>>(rowloss, out);
}
